// RectConv2d_4973572129013
// MI455X (gfx1250) — matrix operand order repaired and checked by reading, not compiled, not run on hardware
//
#include <hip/hip_runtime.h>
#include <math.h>

// ---------------------------------------------------------------------------
// CDNA5 (gfx1250) wave32 WMMA pipeline for the rotated-deformable conv ref.
// All GEMM-shaped work (7x 64->64 3x3 convs + the C*N=1600 sampling einsum)
// runs on v_wmma_f32_16x16x32_f16 with f32 accumulation.
// ---------------------------------------------------------------------------

typedef __attribute__((ext_vector_type(16))) _Float16 v16h;
typedef __attribute__((ext_vector_type(8)))  _Float16 v8h;
typedef __attribute__((ext_vector_type(4)))  _Float16 v4h;
typedef __attribute__((ext_vector_type(8)))  float    v8f;
typedef __attribute__((ext_vector_type(4)))  int      v4i;

#define BATCH 2
#define CC 64          // channels
#define HH 128
#define WW 128
#define HP 130         // padded
#define WP 130
#define KCONV 576      // C*9
#define NPTS 25        // 5x5 sample grid
#define KSAMP 1600     // C*NPTS

// Async global->LDS copy path (gfx1250): guarded so either toolchain compiles.
#if defined(__gfx1250__) && __has_builtin(__builtin_amdgcn_global_load_async_to_lds_b128)
#define HAVE_ASYNC_LDS 1
#else
#define HAVE_ASYNC_LDS 0
#endif

__device__ __forceinline__ void wait_asynccnt0() {
#if defined(__gfx1250__)
#if __has_builtin(__builtin_amdgcn_s_wait_asynccnt)
    __builtin_amdgcn_s_wait_asynccnt(0);
#else
    asm volatile("s_wait_asynccnt 0" ::: "memory");
#endif
#endif
}

__device__ __forceinline__ v16h make_v16(v8h lo, v8h hi) {
    union { v16h v; v8h h[2]; } u;
    u.h[0] = lo; u.h[1] = hi;
    return u.v;
}

__device__ __forceinline__ v8f wmma_f16(v16h a, v16h b, v8f c) {
    // D = A(16x32 f16) * B(32x16 f16) + C(16x16 f32)
    return __builtin_amdgcn_wmma_f32_16x16x32_f16(
        /*neg_a=*/false, a, /*neg_b=*/false, b,
        /*c_mod=*/(short)0, c, /*reuse_a=*/false, /*reuse_b=*/false);
}

// ---------------------------------------------------------------------------
// Packing kernels (one-time per launch, trivially bandwidth bound)
// ---------------------------------------------------------------------------

// x: f32 NCHW [B,64,128,128]  ->  padded NHWC f16 [B,130,130,64]
__global__ __launch_bounds__(256)
void pack_x_kernel(const float* __restrict__ xin, _Float16* __restrict__ xhp) {
    int idx = blockIdx.x * 256 + threadIdx.x;
    const int total = BATCH * HP * WP * CC;
    if (idx >= total) return;
    int c  = idx & 63;
    int xx = (idx >> 6) % WP;
    int yy = (idx / (CC * WP)) % HP;
    int b  = idx / (CC * WP * HP);
    float v = 0.0f;
    if (yy >= 1 && yy <= HH && xx >= 1 && xx <= WW)
        v = xin[(((size_t)b * CC + c) * HH + (yy - 1)) * WW + (xx - 1)];
    xhp[idx] = (_Float16)v;
}

// 3x3 conv weight [64][64][3][3] f32 -> packed f16 [O=64][K=576], K = r*64+c
__global__ __launch_bounds__(256)
void pack_w9_kernel(const float* __restrict__ w, _Float16* __restrict__ wp) {
    int idx = blockIdx.x * 256 + threadIdx.x;
    if (idx >= 64 * KCONV) return;
    int o = idx / KCONV, k = idx % KCONV;
    int r = k >> 6, c = k & 63;
    int ky = r / 3, kx = r % 3;
    wp[idx] = (_Float16)w[(((size_t)o * 64 + c) * 3 + ky) * 3 + kx];
}

// w3 [64][64][25] f32 -> packed f16 [O=64][K=1600], K = n*64+c
__global__ __launch_bounds__(256)
void pack_w3_kernel(const float* __restrict__ w, _Float16* __restrict__ wp) {
    int idx = blockIdx.x * 256 + threadIdx.x;
    if (idx >= 64 * KSAMP) return;
    int o = idx / KSAMP, k = idx % KSAMP;
    int n = k >> 6, c = k & 63;
    wp[idx] = (_Float16)w[((size_t)o * 64 + c) * NPTS + n];
}

// zero a half buffer (grid-stride, 16B chunks)
__global__ __launch_bounds__(256)
void zero_h_kernel(_Float16* __restrict__ p, int n_chunks16B) {
    int4 z = {0, 0, 0, 0};
    for (int i = blockIdx.x * 256 + threadIdx.x; i < n_chunks16B;
         i += gridDim.x * 256)
        reinterpret_cast<int4*>(p)[i] = z;
}

// ---------------------------------------------------------------------------
// 64->64 3x3 conv as implicit GEMM with WMMA.
//   in_p  : padded NHWC f16 [B,130,130,64]
//   wpack : f16 [64][576]
//   out_h : padded NHWC f16 (interior written) or null
//   out_f : f32 NCHW [B,64,128,128] or null
//   act   : 0=relu, 1=sin, 2=none
// grid = B*H*2 blocks (64-pixel strips), 256 threads (8 wave32s)
// ---------------------------------------------------------------------------
__global__ __launch_bounds__(256)
void conv3x3_wmma_kernel(const _Float16* __restrict__ in_p,
                         const _Float16* __restrict__ wpack,
                         const float*    __restrict__ bias,
                         _Float16*       __restrict__ out_h,
                         float*          __restrict__ out_f,
                         int act) {
    __shared__ _Float16 s_in[3 * 66 * CC];   // 3 rows x 66 px x 64 ch = 25,344 B

    const int blk = blockIdx.x;
    const int x0  = (blk & 1) * 64;
    const int y   = (blk >> 1) & 127;
    const int b   = blk >> 8;
    const int tid = threadIdx.x;

    // Cooperative stage of the 3x66x64 f16 input region (b128 chunks).
    const int CHUNKS_PER_ROW = 66 * CC / 8;   // 528
#if HAVE_ASYNC_LDS
    // Async path: GLOBAL_LOAD_ASYNC_TO_LDS_B128, tracked by ASYNCcnt,
    // no VGPR round trip.
    for (int i = tid; i < 3 * CHUNKS_PER_ROW; i += 256) {
        int row = i / CHUNKS_PER_ROW;
        int rem = i - row * CHUNKS_PER_ROW;
        const _Float16* src =
            in_p + (((size_t)b * HP + (y + row)) * WP + x0) * CC + rem * 8;
        _Float16* dst = &s_in[(row * CHUNKS_PER_ROW + rem) * 8];
        __builtin_amdgcn_global_load_async_to_lds_b128(
            (__attribute__((address_space(1))) v4i*)src,
            (__attribute__((address_space(3))) v4i*)dst,
            /*offset=*/0, /*cpol=*/0);
    }
    wait_asynccnt0();
#else
    for (int i = tid; i < 3 * CHUNKS_PER_ROW; i += 256) {
        int row = i / CHUNKS_PER_ROW;
        int rem = i - row * CHUNKS_PER_ROW;
        const int4* src = reinterpret_cast<const int4*>(
            in_p + (((size_t)b * HP + (y + row)) * WP + x0) * CC + rem * 8);
        reinterpret_cast<int4*>(s_in)[row * CHUNKS_PER_ROW + rem] = *src;
    }
#endif
    __syncthreads();

    const int wave = tid >> 5;
    const int lane = tid & 31;
    const int mblk = (wave & 3) * 16;    // 4 pixel blocks of 16
    const int nblk = (wave >> 2) * 32;   // 2 channel blocks of 32
    const int hi   = lane >> 4;          // lane half (K-group selector)
    const int mrow = mblk + (lane & 15); // A-fragment row (pixel)
    const int oA   = nblk + (lane & 15); // B-fragment rows (out channels)
    const int oB   = oA + 16;

    v8f acc0 = {}; v8f acc1 = {};

#pragma unroll
    for (int r = 0; r < 9; ++r) {
        const int ky = r / 3, kx = r % 3;
        const _Float16* abase = s_in + ((ky * 66) + mrow + kx) * CC;
        if (r < 8) {  // pull next weight slice toward the WGP (global_prefetch_b8)
            __builtin_prefetch(wpack + (size_t)oA * KCONV + (r + 1) * 64, 0, 3);
        }
#pragma unroll
        for (int ch = 0; ch < 2; ++ch) {
            const int c0 = ch * 32;
            // A fragment: lanes 0-15 K{0-15}, lanes 16-31 K{16-31}
            const v8h* ap = reinterpret_cast<const v8h*>(abase + c0);
            v16h afrag = make_v16(ap[2 * hi], ap[2 * hi + 1]);
            // B fragments: lane holds out-channel row, 16 contiguous K halves
            const v8h* bp0 = reinterpret_cast<const v8h*>(
                wpack + (size_t)oA * KCONV + r * 64 + c0 + hi * 16);
            const v8h* bp1 = reinterpret_cast<const v8h*>(
                wpack + (size_t)oB * KCONV + r * 64 + c0 + hi * 16);
            v16h bf0 = make_v16(bp0[0], bp0[1]);
            v16h bf1 = make_v16(bp1[0], bp1[1]);
            acc0 = wmma_f16(afrag, bf0, acc0);
            acc1 = wmma_f16(afrag, bf1, acc1);
        }
    }

    // Epilogue: D row M = v + 8*hi, col N = lane&15
    union { v8f v; float f[8]; } u0, u1;
    u0.v = acc0; u1.v = acc1;
    const float bA = bias[oA], bB = bias[oB];
#pragma unroll
    for (int vi = 0; vi < 8; ++vi) {
        const int m = mblk + vi + 8 * hi;
        const int x = x0 + m;
        float vA = u0.f[vi] + bA;
        float vB = u1.f[vi] + bB;
        if (act == 0)      { vA = fmaxf(vA, 0.0f); vB = fmaxf(vB, 0.0f); }
        else if (act == 1) { vA = sinf(vA);        vB = sinf(vB); }
        if (out_h) {
            size_t base = (((size_t)b * HP + (y + 1)) * WP + (x + 1)) * CC;
            out_h[base + oA] = (_Float16)vA;
            out_h[base + oB] = (_Float16)vB;
        }
        if (out_f) {
            out_f[(((size_t)b * CC + oA) * HH + y) * WW + x] = vA;
            out_f[(((size_t)b * CC + oB) * HH + y) * WW + x] = vB;
        }
    }
}

// ---------------------------------------------------------------------------
// Tiny 64->1 head: sin(conv3x3(h)+b2) * w3 + b3  -> sigmoid/softsign map
// mode 0: sigmoid(v)*7+1   (l / w maps, lmax=wmax=8)
// mode 1: softsign(v)*PI   (theta map)
// ---------------------------------------------------------------------------
__global__ __launch_bounds__(256)
void head_kernel(const _Float16* __restrict__ hb,
                 const float* __restrict__ w2, const float* __restrict__ b2,
                 const float* __restrict__ w3, const float* __restrict__ b3,
                 float* __restrict__ outmap, int mode) {
    int idx = blockIdx.x * 256 + threadIdx.x;
    if (idx >= BATCH * HH * WW) return;
    const int x = idx & 127, y = (idx >> 7) & 127, b = idx >> 14;
    float acc = 0.0f;
    for (int ky = 0; ky < 3; ++ky) {
        for (int kx = 0; kx < 3; ++kx) {
            const _Float16* p =
                hb + (((size_t)b * HP + (y + ky)) * WP + (x + kx)) * CC;
#pragma unroll
            for (int c0 = 0; c0 < CC; c0 += 8) {
                v8h h = *reinterpret_cast<const v8h*>(p + c0);
#pragma unroll
                for (int j = 0; j < 8; ++j)
                    acc += (float)h[j] * w2[((c0 + j) * 3 + ky) * 3 + kx];
            }
        }
    }
    float s = sinf(acc + b2[0]);
    float v = s * w3[0] + b3[0];
    float r;
    if (mode == 0) r = 1.0f / (1.0f + expf(-v)) * 7.0f + 1.0f;
    else           r = (v / (1.0f + fabsf(v))) * 3.1415926f;
    outmap[idx] = r;
}

// ---------------------------------------------------------------------------
// Fused rotated bilinear sampling + einsum('bchwn,ocn->bohw') with WMMA.
// Double-buffered LDS A-tile: gather for K-step k+1 overlaps the WMMA of
// step k; one barrier per K-step. Epilogue fuses out3*m + bias.
// grid = B*H*8 (16-pixel strips), 128 threads (4 wave32s, one 16-ch tile each)
// ---------------------------------------------------------------------------
__global__ __launch_bounds__(128)
void sample_gemm_wmma_kernel(const _Float16* __restrict__ xhp,
                             const _Float16* __restrict__ w3p,
                             const float* __restrict__ lmap,
                             const float* __restrict__ wmapv,
                             const float* __restrict__ tmapv,
                             const float* __restrict__ mmap,
                             const float* __restrict__ bmap,
                             float* __restrict__ out) {
    __shared__ float    s_cw[16 * NPTS * 4];             // 4 bilinear weights
    __shared__ int      s_co[16 * NPTS * 4];             // 4 corner half-offsets
    __shared__ alignas(16) _Float16 s_a[2][16 * 32];     // double-buffered A tile

    const int blk = blockIdx.x;
    const int x0  = (blk & 7) * 16;
    const int y   = (blk >> 3) & 127;
    const int b   = blk >> 10;
    const int tid = threadIdx.x;

    // Precompute per-(pixel, sample-point) corner offsets + bilinear weights.
    for (int i = tid; i < 16 * NPTS; i += 128) {
        const int m = i / NPTS, n = i - m * NPTS;
        const int x = x0 + m;
        const int pix = (b * HH + y) * WW + x;
        const float lv = lmap[pix], wv = wmapv[pix], th = tmapv[pix];
        const float ct = cosf(th), st = sinf(th);
        const float pnx = (float)(n / 5) - 2.0f;   // meshgrid 'ij', reshape(-1)
        const float pny = (float)(n % 5) - 2.0f;
        const float px = (float)(y + 1) + (lv * 0.2f) * (pnx * ct - pny * st);
        const float py = (float)(x + 1) + (wv * 0.2f) * (pnx * st + pny * ct);
        const float fx = floorf(px), fy = floorf(py);
        const int qltx = min(max((int)fx,     0), HP - 1);
        const int qlty = min(max((int)fy,     0), WP - 1);
        const int qrbx = min(max((int)fx + 1, 0), HP - 1);
        const int qrby = min(max((int)fy + 1, 0), WP - 1);
        const float pxc = fminf(fmaxf(px, 0.0f), (float)(HP - 1));
        const float pyc = fminf(fmaxf(py, 0.0f), (float)(WP - 1));
        const float dxl = 1.0f + ((float)qltx - pxc);
        const float dxr = 1.0f - ((float)qrbx - pxc);
        const float dyl = 1.0f + ((float)qlty - pyc);
        const float dyr = 1.0f - ((float)qrby - pyc);
        s_cw[i * 4 + 0] = dxl * dyl;  s_co[i * 4 + 0] = (qltx * WP + qlty) * CC;
        s_cw[i * 4 + 1] = dxr * dyr;  s_co[i * 4 + 1] = (qrbx * WP + qrby) * CC;
        s_cw[i * 4 + 2] = dxl * dyr;  s_co[i * 4 + 2] = (qltx * WP + qrby) * CC;
        s_cw[i * 4 + 3] = dxr * dyl;  s_co[i * 4 + 3] = (qrbx * WP + qlty) * CC;
    }

    const int wave = tid >> 5;
    const int lane = tid & 31;
    const int hi   = lane >> 4;
    const int o    = wave * 16 + (lane & 15);  // out channel (B-fragment row)
    const int m_l  = tid >> 3;                 // staging: pixel
    const int cc   = (tid & 7) * 4;            // staging: 4 channels
    const _Float16* xb = xhp + (size_t)b * HP * WP * CC;

    // gather + bilinear blend 4 channels of K-step kk2 into its A buffer
    auto stage = [&](int kk2) {
        const int n  = kk2 >> 1;
        const int c0 = (kk2 & 1) * 32;
        const int e  = (m_l * NPTS + n) * 4;
        float s0 = 0.0f, s1 = 0.0f, s2 = 0.0f, s3 = 0.0f;
#pragma unroll
        for (int j = 0; j < 4; ++j) {
            const float wj = s_cw[e + j];
            const v4h hv = *reinterpret_cast<const v4h*>(
                xb + s_co[e + j] + c0 + cc);
            s0 += wj * (float)hv.x;
            s1 += wj * (float)hv.y;
            s2 += wj * (float)hv.z;
            s3 += wj * (float)hv.w;
        }
        v4h r;
        r.x = (_Float16)s0; r.y = (_Float16)s1;
        r.z = (_Float16)s2; r.w = (_Float16)s3;
        *reinterpret_cast<v4h*>(&s_a[kk2 & 1][m_l * 32 + cc]) = r;
    };

    __syncthreads();          // s_cw / s_co ready
    stage(0);
    __syncthreads();          // buffer 0 ready

    v8f acc = {};
#pragma unroll 2
    for (int kk = 0; kk < 50; ++kk) {          // K = 1600 = 50 * 32
        if (kk + 1 < 50) stage(kk + 1);        // overlaps with this step's WMMA
        const v8h* ap =
            reinterpret_cast<const v8h*>(&s_a[kk & 1][(lane & 15) * 32]);
        v16h afrag = make_v16(ap[2 * hi], ap[2 * hi + 1]);
        const v8h* bp = reinterpret_cast<const v8h*>(
            w3p + (size_t)o * KSAMP + kk * 32 + hi * 16);
        v16h bfrag = make_v16(bp[0], bp[1]);
        acc = wmma_f16(afrag, bfrag, acc);
        __syncthreads();                       // one barrier per K-step
    }

    union { v8f v; float f[8]; } u; u.v = acc;
#pragma unroll
    for (int vi = 0; vi < 8; ++vi) {
        const int m = vi + 8 * hi;
        const int x = x0 + m;
        const size_t idx = (((size_t)b * CC + o) * HH + y) * WW + x;
        out[idx] = u.f[vi] * mmap[idx] + bmap[idx];
    }
}

// ---------------------------------------------------------------------------
// Host-side orchestration
// ---------------------------------------------------------------------------
extern "C" void kernel_launch(void* const* d_in, const int* in_sizes, int n_in,
                              void* d_out, int out_size, void* d_ws, size_t ws_size,
                              hipStream_t stream) {
    (void)in_sizes; (void)n_in; (void)out_size; (void)ws_size;

    const float* x   = (const float*)d_in[0];
    const float* wm1 = (const float*)d_in[1];  const float* bm1 = (const float*)d_in[2];
    const float* wm2 = (const float*)d_in[3];  const float* bm2 = (const float*)d_in[4];
    const float* wb1 = (const float*)d_in[5];  const float* bb1 = (const float*)d_in[6];
    const float* wb2 = (const float*)d_in[7];  const float* bb2 = (const float*)d_in[8];
    const float* wl1 = (const float*)d_in[9];  const float* bl1 = (const float*)d_in[10];
    const float* wl2 = (const float*)d_in[11]; const float* bl2 = (const float*)d_in[12];
    const float* wl3 = (const float*)d_in[13]; const float* bl3 = (const float*)d_in[14];
    const float* ww1 = (const float*)d_in[15]; const float* bw1 = (const float*)d_in[16];
    const float* ww2 = (const float*)d_in[17]; const float* bw2 = (const float*)d_in[18];
    const float* ww3 = (const float*)d_in[19]; const float* bw3 = (const float*)d_in[20];
    const float* wt1 = (const float*)d_in[21]; const float* bt1 = (const float*)d_in[22];
    const float* wt2 = (const float*)d_in[23]; const float* bt2 = (const float*)d_in[24];
    const float* wt3 = (const float*)d_in[25]; const float* bt3 = (const float*)d_in[26];
    const float* w3  = (const float*)d_in[27];

    // Workspace carve-up (all offsets 256B aligned).
    char* ws = (char*)d_ws;
    const size_t XHP_HALVES  = (size_t)BATCH * HP * WP * CC;            // 2,163,200
    const size_t W9_HALVES   = (size_t)64 * KCONV;                      //    36,864
    const size_t W3P_HALVES  = (size_t)64 * KSAMP;                      //   102,400
    const size_t MAP_FLOATS  = (size_t)BATCH * CC * HH * WW;            // 2,097,152
    const size_t SMAP_FLOATS = (size_t)BATCH * HH * WW;                 //    32,768

    size_t off = 0;
    _Float16* xhp  = (_Float16*)(ws + off); off += XHP_HALVES * 2;
    _Float16* hbuf = (_Float16*)(ws + off); off += XHP_HALVES * 2;
    _Float16* wp[7];
    for (int i = 0; i < 7; ++i) { wp[i] = (_Float16*)(ws + off); off += W9_HALVES * 2; }
    _Float16* w3p  = (_Float16*)(ws + off); off += W3P_HALVES * 2;
    float* mmap    = (float*)(ws + off); off += MAP_FLOATS * 4;
    float* bmap    = (float*)(ws + off); off += MAP_FLOATS * 4;
    float* lmap    = (float*)(ws + off); off += SMAP_FLOATS * 4;
    float* wmapv   = (float*)(ws + off); off += SMAP_FLOATS * 4;
    float* tmapv   = (float*)(ws + off); off += SMAP_FLOATS * 4;

    // 0. zero hidden buffer (its halo must be 0 for the second convs)
    zero_h_kernel<<<1024, 256, 0, stream>>>(hbuf, (int)(XHP_HALVES / 8));

    // 1. pack x -> padded NHWC f16
    pack_x_kernel<<<(int)((XHP_HALVES + 255) / 256), 256, 0, stream>>>(x, xhp);

    // 2. pack 3x3 weights + w3
    const int g9 = (int)((W9_HALVES + 255) / 256);
    pack_w9_kernel<<<g9, 256, 0, stream>>>(wm1, wp[0]);
    pack_w9_kernel<<<g9, 256, 0, stream>>>(wm2, wp[1]);
    pack_w9_kernel<<<g9, 256, 0, stream>>>(wb1, wp[2]);
    pack_w9_kernel<<<g9, 256, 0, stream>>>(wb2, wp[3]);
    pack_w9_kernel<<<g9, 256, 0, stream>>>(wl1, wp[4]);
    pack_w9_kernel<<<g9, 256, 0, stream>>>(ww1, wp[5]);
    pack_w9_kernel<<<g9, 256, 0, stream>>>(wt1, wp[6]);
    pack_w3_kernel<<<(int)((W3P_HALVES + 255) / 256), 256, 0, stream>>>(w3, w3p);

    const int gconv = BATCH * HH * 2;   // 512 workgroups
    const int ghead = (int)((SMAP_FLOATS + 255) / 256);

    // 3. m = conv(relu(conv(x)))
    conv3x3_wmma_kernel<<<gconv, 256, 0, stream>>>(xhp,  wp[0], bm1, hbuf, nullptr, 0);
    conv3x3_wmma_kernel<<<gconv, 256, 0, stream>>>(hbuf, wp[1], bm2, nullptr, mmap, 2);
    // 4. bias = conv(relu(conv(x)))
    conv3x3_wmma_kernel<<<gconv, 256, 0, stream>>>(xhp,  wp[2], bb1, hbuf, nullptr, 0);
    conv3x3_wmma_kernel<<<gconv, 256, 0, stream>>>(hbuf, wp[3], bb2, nullptr, bmap, 2);
    // 5. l / w / theta heads: sin(conv) -> 64->1 head
    conv3x3_wmma_kernel<<<gconv, 256, 0, stream>>>(xhp, wp[4], bl1, hbuf, nullptr, 1);
    head_kernel<<<ghead, 256, 0, stream>>>(hbuf, wl2, bl2, wl3, bl3, lmap, 0);
    conv3x3_wmma_kernel<<<gconv, 256, 0, stream>>>(xhp, wp[5], bw1, hbuf, nullptr, 1);
    head_kernel<<<ghead, 256, 0, stream>>>(hbuf, ww2, bw2, ww3, bw3, wmapv, 0);
    conv3x3_wmma_kernel<<<gconv, 256, 0, stream>>>(xhp, wp[6], bt1, hbuf, nullptr, 1);
    head_kernel<<<ghead, 256, 0, stream>>>(hbuf, wt2, bt2, wt3, bt3, tmapv, 1);

    // 6. fused rotated bilinear sampling + einsum + (*m + bias)
    sample_gemm_wmma_kernel<<<BATCH * HH * 8, 128, 0, stream>>>(
        xhp, w3p, lmap, wmapv, tmapv, mmap, bmap, (float*)d_out);
}
